// SigLoss_18597208391789
// MI455X (gfx1250) — compile-verified, matrix operand order checked by reading, not run on hardware
//
#include <hip/hip_runtime.h>
#include <hip/hip_bf16.h>

#ifndef __has_builtin
#define __has_builtin(x) 0
#endif

typedef __attribute__((ext_vector_type(16))) _Float16 v16h;
typedef __attribute__((ext_vector_type(8)))  _Float16 v8h;
typedef __attribute__((ext_vector_type(4)))  _Float16 v4h;
typedef __attribute__((ext_vector_type(8)))  float    v8f;
typedef int v4i_vs __attribute__((vector_size(16)));   // matches builtin's pointee type

#define NB 8192
#define ND 1024

#define AS1 __attribute__((address_space(1)))
#define AS3 __attribute__((address_space(3)))

// LDS staging geometry: block tile 256(M) x 128(N), K-step 32 halves (64B/row),
// rows padded to 80B (20-bank stride -> conflict-free ds_load_b128, 16B aligned).
#define ROW_BYTES   80
#define A_ROWS      256
#define B_ROWS      128
#define A_BYTES     (A_ROWS * ROW_BYTES)   // 20480
#define B_BYTES     (B_ROWS * ROW_BYTES)   // 10240
#define STAGE_BYTES (A_BYTES + B_BYTES)    // 30720
#define NKSTEP      (ND / 32)              // 32

// ---- CDNA5 async global->LDS copy (ASYNCcnt-tracked) ----
__device__ __forceinline__ void async_ld_b128(void* lds_ptr, const void* gptr) {
#if __has_builtin(__builtin_amdgcn_global_load_async_to_lds_b128)
    __builtin_amdgcn_global_load_async_to_lds_b128(
        (AS1 v4i_vs*)gptr, (AS3 v4i_vs*)lds_ptr, 0, 0);
#else
    asm volatile("global_load_async_to_lds_b128 %0, %1, off"
                 :: "v"((AS3 void*)lds_ptr), "v"(gptr) : "memory");
#endif
}

#if __has_builtin(__builtin_amdgcn_s_wait_asynccnt)
#define WAIT_ASYNC(n) __builtin_amdgcn_s_wait_asynccnt(n)
#else
#define WAIT_ASYNC(n) asm volatile("s_wait_asynccnt %0" :: "i"(n) : "memory")
#endif

// ---------------- zero the loss accumulator ----------------
__global__ void sig_init(float* __restrict__ acc) {
    if (threadIdx.x == 0 && blockIdx.x == 0) acc[0] = 0.f;
}

// ---------------- L2-normalize rows + convert f32 -> f16 ----------------
// One wave (32 lanes) per row; 16384 rows total (img then txt).
__global__ __launch_bounds__(256) void sig_norm_cvt(
        const float* __restrict__ img, const float* __restrict__ txt,
        _Float16* __restrict__ img16, _Float16* __restrict__ txt16) {
    const int gwave = (blockIdx.x * 256 + threadIdx.x) >> 5;
    const int lane  = threadIdx.x & 31;
    const float* src;
    _Float16*    dst;
    int row;
    if (gwave < NB) { src = img; dst = img16; row = gwave; }
    else            { src = txt; dst = txt16; row = gwave - NB; }

    const float4* p = (const float4*)(src + (size_t)row * ND);
    float4 v[8];
    float ss = 0.f;
#pragma unroll
    for (int j = 0; j < 8; ++j) {
        float4 q = p[j * 32 + lane];   // coalesced 128-float chunks
        v[j] = q;
        ss += q.x*q.x + q.y*q.y + q.z*q.z + q.w*q.w;
    }
#pragma unroll
    for (int off = 16; off >= 1; off >>= 1) ss += __shfl_xor(ss, off, 32);
    const float scale = 1.f / fmaxf(sqrtf(ss), 1e-12f);   // matches F.normalize eps

    _Float16* drow = dst + (size_t)row * ND;
#pragma unroll
    for (int j = 0; j < 8; ++j) {
        v4h h;
        h[0] = (_Float16)(v[j].x * scale);
        h[1] = (_Float16)(v[j].y * scale);
        h[2] = (_Float16)(v[j].z * scale);
        h[3] = (_Float16)(v[j].w * scale);
        *(v4h*)(drow + j * 128 + lane * 4) = h;           // 8B coalesced store
    }
}

// Assemble a 16x32 f16 fragment from a staged LDS row.
// Per-lane: halves 0..7 at byte koff, halves 8..15 at byte koff+32.
__device__ __forceinline__ v16h lds_frag(const char* rowp, int koff) {
    v8h lo = *(const v8h*)(rowp + koff);        // ds_load_b128
    v8h hi = *(const v8h*)(rowp + koff + 32);   // ds_load_b128
    return __builtin_shufflevector(lo, hi, 0,1,2,3,4,5,6,7,8,9,10,11,12,13,14,15);
}

// Issue one K-step stage: A 256 rows + B 128 rows, 64B per row, via async DMA.
// Each of 256 threads issues 6 x b128 (A:4, B:2) -> ASYNCcnt += 6 per wave.
__device__ __forceinline__ void stage_tiles(
        char* buf, const _Float16* __restrict__ A, const _Float16* __restrict__ Bt,
        int m_blk, int n_blk, int k0, int tid) {
    const int seg   = tid & 3;        // 16B segment within the 64B row
    const int rbase = tid >> 2;       // 0..63
#pragma unroll
    for (int i = 0; i < 4; ++i) {
        const int row = i * 64 + rbase;
        async_ld_b128(buf + row * ROW_BYTES + seg * 16,
                      A + (size_t)(m_blk + row) * ND + k0 + seg * 8);
    }
#pragma unroll
    for (int i = 0; i < 2; ++i) {
        const int row = i * 64 + rbase;
        async_ld_b128(buf + A_BYTES + row * ROW_BYTES + seg * 16,
                      Bt + (size_t)(n_blk + row) * ND + k0 + seg * 8);
    }
}

// ---------------- fused GEMM + sigmoid-loss epilogue ----------------
// Block tile 256(M) x 128(N); 8 waves as 4(M) x 2(N); wave tile 64x64 = 4x4 WMMA tiles.
__global__ __launch_bounds__(256) void sig_gemm(
        const _Float16* __restrict__ A,   // img16 [NB][ND]
        const _Float16* __restrict__ Bt,  // txt16 [NB][ND]  (rows of B^T)
        const float* __restrict__ t_prime_p,
        const float* __restrict__ bias_p,
        float* __restrict__ acc_out) {
    __shared__ alignas(16) char lds[2][STAGE_BYTES];      // 60 KB double buffer

    const int tid  = threadIdx.x;
    const int lane = tid & 31;
    const int wave = tid >> 5;
    const int wm = wave & 3;
    const int wn = wave >> 2;
    const int m_blk = blockIdx.x * 256;
    const int n_blk = blockIdx.y * 128;
    const int m_base = m_blk + wm * 64;
    const int n_base = n_blk + wn * 64;
    const int l15  = lane & 15;
    const int koff = (lane >> 4) << 4;    // byte offset 0 or 16 within staged row

    v8f acc[4][4];
#pragma unroll
    for (int i = 0; i < 4; ++i)
#pragma unroll
        for (int j = 0; j < 4; ++j)
#pragma unroll
            for (int e = 0; e < 8; ++e) acc[i][j][e] = 0.f;

    // Prologue: stage K-step 0 into buffer 0.
    stage_tiles(&lds[0][0], A, Bt, m_blk, n_blk, 0, tid);

    for (int ks = 0; ks < NKSTEP; ++ks) {
        char* cbuf = &lds[ks & 1][0];
        if (ks + 1 < NKSTEP) {
            stage_tiles(&lds[(ks + 1) & 1][0], A, Bt, m_blk, n_blk, (ks + 1) * 32, tid);
            WAIT_ASYNC(6);    // 6 newest belong to stage ks+1 -> stage ks complete
        } else {
            WAIT_ASYNC(0);
        }
        __syncthreads();      // publish staged tile to all waves

        v16h af[4], bf[4];
#pragma unroll
        for (int t = 0; t < 4; ++t)
            af[t] = lds_frag(cbuf + (wm * 64 + t * 16 + l15) * ROW_BYTES, koff);
#pragma unroll
        for (int t = 0; t < 4; ++t)
            bf[t] = lds_frag(cbuf + A_BYTES + (wn * 64 + t * 16 + l15) * ROW_BYTES, koff);
#pragma unroll
        for (int i = 0; i < 4; ++i)
#pragma unroll
            for (int j = 0; j < 4; ++j)
                acc[i][j] = __builtin_amdgcn_wmma_f32_16x16x32_f16(
                    false, af[i], false, bf[j], (short)0, acc[i][j], false, false);

        __syncthreads();      // all waves done reading before next stage overwrites
    }

    // Epilogue: logits = acc*t + bias; labels = +1 diag, -1 off-diag;
    // log_sigmoid(x) = min(x,0) - log1p(exp(-|x|))
    const float t_scale = fminf(__expf(t_prime_p[0]), 100.f);
    const float bias    = bias_p[0];
    const int row_off = (lane >> 4) << 3;  // C/D layout: elem e -> M = e + 8*(lane/16)
    const int col     = lane & 15;         //                N = lane % 16
    float sum = 0.f;
#pragma unroll
    for (int i = 0; i < 4; ++i)
#pragma unroll
        for (int j = 0; j < 4; ++j) {
            const int mb = m_base + i * 16 + row_off;
            const int n  = n_base + j * 16 + col;
#pragma unroll
            for (int e = 0; e < 8; ++e) {
                float logit = acc[i][j][e] * t_scale + bias;
                float x = ((mb + e) == n) ? logit : -logit;
                sum += fminf(x, 0.f) - log1pf(__expf(-fabsf(x)));
            }
        }

#pragma unroll
    for (int off = 16; off >= 1; off >>= 1) sum += __shfl_xor(sum, off, 32);
    __shared__ float wsum[8];
    if (lane == 0) wsum[wave] = sum;
    __syncthreads();
    if (tid == 0) {
        float s = 0.f;
#pragma unroll
        for (int i = 0; i < 8; ++i) s += wsum[i];
        atomicAdd(acc_out, s);   // one atomic per block (2048 total)
    }
}

__global__ void sig_final(const float* __restrict__ acc, float* __restrict__ out) {
    if (threadIdx.x == 0 && blockIdx.x == 0) out[0] = -acc[0] / (float)NB;
}

extern "C" void kernel_launch(void* const* d_in, const int* in_sizes, int n_in,
                              void* d_out, int out_size, void* d_ws, size_t ws_size,
                              hipStream_t stream) {
    (void)in_sizes; (void)n_in; (void)out_size; (void)ws_size;
    const float* img     = (const float*)d_in[0];
    const float* txt     = (const float*)d_in[1];
    const float* t_prime = (const float*)d_in[2];
    const float* bias    = (const float*)d_in[3];
    float* out = (float*)d_out;

    char* ws = (char*)d_ws;
    float*    acc   = (float*)ws;                          // 1 float (padded to 256B)
    _Float16* img16 = (_Float16*)(ws + 256);               // 16 MB
    _Float16* txt16 = img16 + (size_t)NB * ND;             // 16 MB

    sig_init<<<1, 1, 0, stream>>>(acc);
    sig_norm_cvt<<<(2 * NB) / 8, 256, 0, stream>>>(img, txt, img16, txt16);
    dim3 grid(NB / 256, NB / 128);                         // 32 x 64
    sig_gemm<<<grid, 256, 0, stream>>>(img16, txt16, t_prime, bias, acc);
    sig_final<<<1, 1, 0, stream>>>(acc, out);
}
